// HierarchicalGNNBlock_42279658062317
// MI455X (gfx1250) — compile-verified, matrix operand order checked by reading, not run on hardware
//
#include <hip/hip_runtime.h>
#include <math.h>

// ---- problem constants (match reference) ----
#define NN 100000
#define EE 400000
#define SS 1024
#define EBE 400000
#define ESE 50000
#define LF 128      // feature dim
#define HF 256      // hidden dim
#define KC 384      // 3*LF concat dim

typedef __attribute__((ext_vector_type(16))) __bf16 v16bf;
typedef __attribute__((ext_vector_type(8)))  __bf16 v8bf;
typedef __attribute__((ext_vector_type(4)))  __bf16 v4bf;
typedef __attribute__((ext_vector_type(8)))  float  v8f;

// A/B-fragment local-K mapping for v_wmma_*_16x16x32 (bf16), ISA 7.12.2:
// K_local(j,lane) = j + 8*(j>=8) + 8*(lane>=16)
__device__ __forceinline__ int k_local(int j, int lane) {
  return j + ((j >= 8) ? 8 : 0) + ((lane >= 16) ? 8 : 0);
}

// ---------------- utility kernels ----------------

__global__ void zero_f32_kernel(float* __restrict__ p, int n) {
  int i = blockIdx.x * blockDim.x + threadIdx.x;
  if (i < n) p[i] = 0.f;
}

// Pack W1 (4,384,256) fp32 -> bf16 in WMMA fragment order [mlp][cb:16][kc:12][lane:32][j:16]
__global__ void pack_w1_kernel(const float* __restrict__ W, __bf16* __restrict__ Wp) {
  int i = blockIdx.x * blockDim.x + threadIdx.x;
  if (i >= 4 * 16 * 12 * 32 * 16) return;
  int j    = i & 15;
  int lane = (i >> 4) & 31;
  int t    = i >> 9;            // mlp*192 + cb*12 + kc
  int kc   = t % 12;
  int cb   = (t / 12) % 16;
  int mlp  = t / (12 * 16);
  int k = kc * 32 + k_local(j, lane);
  int n = cb * 16 + (lane & 15);
  Wp[i] = (__bf16)W[((size_t)mlp * KC + k) * HF + n];
}

// Pack W2 (4,256,128) fp32 -> bf16 in fragment order [mlp][cb:8][kc:8][lane:32][j:16]
__global__ void pack_w2_kernel(const float* __restrict__ W, __bf16* __restrict__ Wp) {
  int i = blockIdx.x * blockDim.x + threadIdx.x;
  if (i >= 4 * 8 * 8 * 32 * 16) return;
  int j    = i & 15;
  int lane = (i >> 4) & 31;
  int t    = i >> 9;            // mlp*64 + cb*8 + kc
  int kc   = t % 8;
  int cb   = (t / 8) % 8;
  int mlp  = t / (8 * 8);
  int k = kc * 32 + k_local(j, lane);
  int n = cb * 16 + (lane & 15);
  Wp[i] = (__bf16)W[((size_t)mlp * HF + k) * LF + n];
}

// out[idx_dst[e]][:] += w[e] * src[idx_src[e] or e][:]   (32 threads/edge, 4 floats/thread)
__global__ void scatter_wsum_kernel(const float* __restrict__ src,
                                    const float* __restrict__ w,
                                    const int* __restrict__ idx_src,
                                    const int* __restrict__ idx_dst,
                                    float* __restrict__ out, int nE) {
  int t = blockIdx.x * blockDim.x + threadIdx.x;
  int e = t >> 5;
  int c = (t & 31) << 2;
  if (e >= nE) return;
  float we = w[e];
  int s = idx_src ? idx_src[e] : e;
  int d = idx_dst[e];
  const float4 v = *(const float4*)(src + (size_t)s * LF + c);
  float* o = out + (size_t)d * LF + c;
  atomicAdd(o + 0, we * v.x);
  atomicAdd(o + 1, we * v.y);
  atomicAdd(o + 2, we * v.z);
  atomicAdd(o + 3, we * v.w);
}

// load one A fragment (16x32 bf16) for row m from an LDS tile row (two 16B loads)
template <int STRIDE>
__device__ __forceinline__ v16bf load_a_frag(const __bf16* tile, int m, int kc, int lane) {
  const int hi = (lane >= 16) ? 8 : 0;
  union { v16bf v; v8bf h[2]; } au;
  au.h[0] = *(const v8bf*)(tile + (size_t)m * STRIDE + kc * 32 + hi);
  au.h[1] = *(const v8bf*)(tile + (size_t)m * STRIDE + kc * 32 + 16 + hi);
  return au.v;
}

// ---------------- fused concat -> GEMM(384x256) -> +bias -> LN -> ReLU -> bf16 hidden ----------------
// 512 threads = 16 waves; block computes 32 rows x 256 cols.
// Wave w owns cols [16w,16w+16) for both 16-row halves (B fragment reused twice).
__global__ void __launch_bounds__(512)
mlp_layer1_kernel(const float* __restrict__ s0, const int* __restrict__ i0,
                  const float* __restrict__ s1, const int* __restrict__ i1,
                  const float* __restrict__ s2, const int* __restrict__ i2,
                  const __bf16* __restrict__ Wp, const float* __restrict__ bias,
                  const float* __restrict__ gamma, const float* __restrict__ beta,
                  __bf16* __restrict__ hid, int M) {
  __shared__ __align__(16) __bf16 Atile[32][KC];   // 24 KB
  __shared__ __align__(16) float  Ctile[32][HF];   // 32 KB

  const int tid  = threadIdx.x;
  const int lane = tid & 31;
  const int wave = tid >> 5;          // 0..15
  const int row0 = blockIdx.x * 32;

  // stage gathered+concatenated rows into LDS as bf16 (float4 -> packed 4x bf16)
  for (int q = tid; q < 32 * (KC / 4); q += 512) {
    int r = q / (KC / 4);
    int c = (q - r * (KC / 4)) * 4;
    int gr = row0 + r;
    float4 v = make_float4(0.f, 0.f, 0.f, 0.f);
    if (gr < M) {
      if (c < LF)            v = *(const float4*)(s0 + (size_t)(i0 ? i0[gr] : gr) * LF + c);
      else if (c < 2 * LF)   v = *(const float4*)(s1 + (size_t)(i1 ? i1[gr] : gr) * LF + (c - LF));
      else                   v = *(const float4*)(s2 + (size_t)(i2 ? i2[gr] : gr) * LF + (c - 2 * LF));
    }
    v4bf p;
    p[0] = (__bf16)v.x; p[1] = (__bf16)v.y; p[2] = (__bf16)v.z; p[3] = (__bf16)v.w;
    *(v4bf*)&Atile[r][c] = p;
  }
  __syncthreads();

  // WMMA K-loop (EXEC all ones). B fragment = contiguous 32B per lane from packed weights.
  v8f acc0 = {}, acc1 = {};
  const int n = lane & 15;
  const int colbase = wave * 16;
#pragma unroll
  for (int kc = 0; kc < KC / 32; ++kc) {
    v16bf b = *(const v16bf*)(Wp + (((size_t)wave * (KC / 32) + kc) * 32 + lane) * 16);
    v16bf a0 = load_a_frag<KC>(&Atile[0][0], n,      kc, lane);
    v16bf a1 = load_a_frag<KC>(&Atile[0][0], n + 16, kc, lane);
    acc0 = __builtin_amdgcn_wmma_f32_16x16x32_bf16(false, a0, false, b, (short)0, acc0, false, false);
    acc1 = __builtin_amdgcn_wmma_f32_16x16x32_bf16(false, a1, false, b, (short)0, acc1, false, false);
  }

  const int rhalf = (lane >= 16) ? 8 : 0;
  const float bv = bias[colbase + n];
#pragma unroll
  for (int r = 0; r < 8; ++r) {
    Ctile[r + rhalf][colbase + n]      = acc0[r] + bv;
    Ctile[16 + r + rhalf][colbase + n] = acc1[r] + bv;
  }
  __syncthreads();

  // LayerNorm + ReLU: one wave per row; each wave handles rows {w, w+16}
#pragma unroll
  for (int half = 0; half < 2; ++half) {
    int r  = wave + half * 16;
    int c0 = lane;
    float s = 0.f, sq = 0.f;
#pragma unroll
    for (int k = 0; k < HF / 32; ++k) {
      float v = Ctile[r][c0 + 32 * k];
      s += v; sq += v * v;
    }
#pragma unroll
    for (int off = 16; off >= 1; off >>= 1) {
      s  += __shfl_xor(s,  off, 32);
      sq += __shfl_xor(sq, off, 32);
    }
    float mean = s * (1.f / HF);
    float var  = sq * (1.f / HF) - mean * mean;
    float rstd = rsqrtf(var + 1e-5f);
    int gr = row0 + r;
    if (gr < M) {
#pragma unroll
      for (int k = 0; k < HF / 32; ++k) {
        int c = c0 + 32 * k;
        float v = (Ctile[r][c] - mean) * rstd * gamma[c] + beta[c];
        hid[(size_t)gr * HF + c] = (__bf16)fmaxf(v, 0.f);
      }
    }
  }
}

// ---------------- GEMM(256x128) -> +bias -> LN -> act -> +residual -> fp32 out ----------------
// 256 threads = 8 waves; block computes 32 rows x 128 cols.
__global__ void __launch_bounds__(256)
mlp_layer2_kernel(const __bf16* __restrict__ hid,
                  const __bf16* __restrict__ Wp, const float* __restrict__ bias,
                  const float* __restrict__ gamma, const float* __restrict__ beta,
                  const float* __restrict__ resid, float* __restrict__ out,
                  int M, int act_tanh) {
  __shared__ __align__(16) __bf16 Atile[32][HF];   // 16 KB
  __shared__ __align__(16) float  Ctile[32][LF];   // 16 KB

  const int tid  = threadIdx.x;
  const int lane = tid & 31;
  const int wave = tid >> 5;   // 0..7
  const int row0 = blockIdx.x * 32;

  // stage hidden tile: 16-byte chunks (8 bf16 per chunk)
  for (int q = tid; q < 32 * (HF / 8); q += 256) {
    int r = q / (HF / 8);
    int c = (q - r * (HF / 8)) * 8;
    int gr = row0 + r;
    uint4 u = make_uint4(0u, 0u, 0u, 0u);
    if (gr < M) u = *(const uint4*)(hid + (size_t)gr * HF + c);
    *(uint4*)&Atile[r][c] = u;
  }
  __syncthreads();

  v8f acc0 = {}, acc1 = {};
  const int n = lane & 15;
  const int colbase = wave * 16;
#pragma unroll
  for (int kc = 0; kc < HF / 32; ++kc) {
    v16bf b = *(const v16bf*)(Wp + (((size_t)wave * (HF / 32) + kc) * 32 + lane) * 16);
    v16bf a0 = load_a_frag<HF>(&Atile[0][0], n,      kc, lane);
    v16bf a1 = load_a_frag<HF>(&Atile[0][0], n + 16, kc, lane);
    acc0 = __builtin_amdgcn_wmma_f32_16x16x32_bf16(false, a0, false, b, (short)0, acc0, false, false);
    acc1 = __builtin_amdgcn_wmma_f32_16x16x32_bf16(false, a1, false, b, (short)0, acc1, false, false);
  }

  const int rhalf = (lane >= 16) ? 8 : 0;
  const float bv = bias[colbase + n];
#pragma unroll
  for (int r = 0; r < 8; ++r) {
    Ctile[r + rhalf][colbase + n]      = acc0[r] + bv;
    Ctile[16 + r + rhalf][colbase + n] = acc1[r] + bv;
  }
  __syncthreads();

  // LN over 128: 16 threads per row; each 16-lane group handles rows {g, g+16}
#pragma unroll
  for (int half = 0; half < 2; ++half) {
    int r  = (tid >> 4) + half * 16;
    int c0 = tid & 15;
    float s = 0.f, sq = 0.f;
#pragma unroll
    for (int k = 0; k < LF / 16; ++k) {
      float v = Ctile[r][c0 + 16 * k];
      s += v; sq += v * v;
    }
#pragma unroll
    for (int off = 8; off >= 1; off >>= 1) {
      s  += __shfl_xor(s,  off, 32);
      sq += __shfl_xor(sq, off, 32);
    }
    float mean = s * (1.f / LF);
    float var  = sq * (1.f / LF) - mean * mean;
    float rstd = rsqrtf(var + 1e-5f);
    int gr = row0 + r;
    if (gr < M) {
#pragma unroll
      for (int k = 0; k < LF / 16; ++k) {
        int c = c0 + 16 * k;
        float v = (Ctile[r][c] - mean) * rstd * gamma[c] + beta[c];
        v = act_tanh ? tanhf(v) : fmaxf(v, 0.f);
        out[(size_t)gr * LF + c] = v + resid[(size_t)gr * LF + c];
      }
    }
  }
}

// ---------------- host launch ----------------

extern "C" void kernel_launch(void* const* d_in, const int* in_sizes, int n_in,
                              void* d_out, int out_size, void* d_ws, size_t ws_size,
                              hipStream_t stream) {
  (void)in_sizes; (void)n_in; (void)out_size; (void)ws_size;

  const float* nodes      = (const float*)d_in[0];
  const float* edges      = (const float*)d_in[1];
  const float* supernodes = (const float*)d_in[2];
  const float* superedges = (const float*)d_in[3];
  const float* ew         = (const float*)d_in[4];
  const float* bw         = (const float*)d_in[5];
  const float* sw         = (const float*)d_in[6];
  const float* W1         = (const float*)d_in[7];
  const float* b1         = (const float*)d_in[8];
  const float* g1         = (const float*)d_in[9];
  const float* be1        = (const float*)d_in[10];
  const float* W2         = (const float*)d_in[11];
  const float* b2         = (const float*)d_in[12];
  const float* g2         = (const float*)d_in[13];
  const float* be2        = (const float*)d_in[14];
  const int*   g0         = (const int*)d_in[15];          // graph[0]
  const int*   g1i        = (const int*)d_in[15] + EE;     // graph[1]
  const int*   bg0        = (const int*)d_in[16];          // bipartite[0] (node)
  const int*   bg1        = (const int*)d_in[16] + EBE;    // bipartite[1] (supernode)
  const int*   sg0        = (const int*)d_in[17];          // super_graph[0]
  const int*   sg1        = (const int*)d_in[17] + ESE;    // super_graph[1]

  // d_out layout: nodes | edges | supernodes | superedges
  float* nodes_out = (float*)d_out;
  float* edges_out = nodes_out + (size_t)NN * LF;
  float* sn_out    = edges_out + (size_t)EE * LF;
  float* se_out    = sn_out    + (size_t)SS * LF;

  // workspace layout
  char* ws = (char*)d_ws;
  __bf16* W1b = (__bf16*)ws;                          ws += (size_t)4 * KC * HF * 2;   // 768 KB
  __bf16* W2b = (__bf16*)ws;                          ws += (size_t)4 * HF * LF * 2;   // 256 KB
  float*  msg0 = (float*)ws;                          ws += (size_t)NN * LF * 4;       // 51.2 MB
  float*  msg1 = (float*)ws;                          ws += (size_t)NN * LF * 4;       // 51.2 MB
  __bf16* hid  = (__bf16*)ws;                         /* E*HF bf16 = 204.8 MB */

  pack_w1_kernel<<<(4 * KC * HF + 511) / 512, 512, 0, stream>>>(W1, W1b);
  pack_w2_kernel<<<(4 * HF * LF + 511) / 512, 512, 0, stream>>>(W2, W2b);

  // ---- stage 0: supernode update ----
  const int nS = SS * LF;
  zero_f32_kernel<<<(nS + 255) / 256, 256, 0, stream>>>(msg0, nS);  // node_msg
  zero_f32_kernel<<<(nS + 255) / 256, 256, 0, stream>>>(msg1, nS);  // attn_msg
  scatter_wsum_kernel<<<(EBE * 32 + 255) / 256, 256, 0, stream>>>(nodes, bw, bg0, bg1, msg0, EBE);
  scatter_wsum_kernel<<<(ESE * 32 + 255) / 256, 256, 0, stream>>>(superedges, sw, sg0, sg1, msg1, ESE);
  {
    int M = SS, blocks = (M + 31) / 32;
    mlp_layer1_kernel<<<blocks, 512, 0, stream>>>(
        supernodes, nullptr, msg1, nullptr, msg0, nullptr,
        W1b + (size_t)0 * KC * HF, b1 + 0 * HF, g1 + 0 * HF, be1 + 0 * HF, hid, M);
    mlp_layer2_kernel<<<blocks, 256, 0, stream>>>(
        hid, W2b + (size_t)0 * HF * LF, b2 + 0 * LF, g2 + 0 * LF, be2 + 0 * LF,
        supernodes, sn_out, M, /*tanh=*/0);
  }

  // ---- stage 1: node update (uses UPDATED supernodes from d_out) ----
  const int nN = NN * LF;
  zero_f32_kernel<<<(nN + 255) / 256, 256, 0, stream>>>(msg0, nN);  // sn_msg
  zero_f32_kernel<<<(nN + 255) / 256, 256, 0, stream>>>(msg1, nN);  // edge_msg
  scatter_wsum_kernel<<<(EBE * 32 + 255) / 256, 256, 0, stream>>>(sn_out, bw, bg1, bg0, msg0, EBE);
  scatter_wsum_kernel<<<(EE * 32 + 255) / 256, 256, 0, stream>>>(edges, ew, nullptr, g1i, msg1, EE);
  {
    int M = NN, blocks = (M + 31) / 32;
    mlp_layer1_kernel<<<blocks, 512, 0, stream>>>(
        nodes, nullptr, msg1, nullptr, msg0, nullptr,
        W1b + (size_t)1 * KC * HF, b1 + 1 * HF, g1 + 1 * HF, be1 + 1 * HF, hid, M);
    mlp_layer2_kernel<<<blocks, 256, 0, stream>>>(
        hid, W2b + (size_t)1 * HF * LF, b2 + 1 * LF, g2 + 1 * LF, be2 + 1 * LF,
        nodes, nodes_out, M, /*tanh=*/0);
  }

  // ---- stage 2: superedge update ----
  {
    int M = ESE, blocks = (M + 31) / 32;
    mlp_layer1_kernel<<<blocks, 512, 0, stream>>>(
        sn_out, sg0, sn_out, sg1, superedges, nullptr,
        W1b + (size_t)2 * KC * HF, b1 + 2 * HF, g1 + 2 * HF, be1 + 2 * HF, hid, M);
    mlp_layer2_kernel<<<blocks, 256, 0, stream>>>(
        hid, W2b + (size_t)2 * HF * LF, b2 + 2 * LF, g2 + 2 * LF, be2 + 2 * LF,
        superedges, se_out, M, /*tanh=*/1);
  }

  // ---- stage 3: edge update ----
  {
    int M = EE, blocks = (M + 31) / 32;
    mlp_layer1_kernel<<<blocks, 512, 0, stream>>>(
        nodes_out, g0, nodes_out, g1i, edges, nullptr,
        W1b + (size_t)3 * KC * HF, b1 + 3 * HF, g1 + 3 * HF, be1 + 3 * HF, hid, M);
    mlp_layer2_kernel<<<blocks, 256, 0, stream>>>(
        hid, W2b + (size_t)3 * HF * LF, b2 + 3 * LF, g2 + 3 * LF, be2 + 3 * LF,
        edges, edges_out, M, /*tanh=*/1);
  }
}
